// GraphConvolutionSparse_53240414601890
// MI455X (gfx1250) — matrix operand order repaired and checked by reading, not compiled, not run on hardware
//
#include <hip/hip_runtime.h>

// ---------------------------------------------------------------------------
// out[16384][128] = adj[16384][16384] @ (inputs[16384][512] @ W[512][128])
//
// MI455X roofline: adj (1 GiB f32) must stream from HBM once -> floor ~46us
// @ 23.3 TB/s (AI fixed at 64 FLOP/B for the big GEMM). bf16 WMMA
// (16x16x32) puts the 68.7 GFLOP under that floor (~30us); f32 WMMA would
// be ~5x compute-bound. Strategy: stream adj f32 -> cvt_pk_bf16 -> LDS,
// accumulate f32 with v_wmma_f32_16x16x32_bf16.
//
// GEMM2 blocking: 64 adj rows x all 128 cols per 256-thread block (8 waves,
// 1 N-tile each, 4 M-subtiles each). Double-buffered LDS + register
// prefetch: next adj tile's global_load_b128s issue before the current
// tile's WMMAs, one barrier per K-step. x (4 MB bf16, transposed) stays
// L2-resident; 64-row tiles cut its re-read to ~1 GB of L2 traffic.
// ---------------------------------------------------------------------------

typedef __attribute__((ext_vector_type(16))) __bf16 bf16x16;
typedef __attribute__((ext_vector_type(8)))  __bf16 bf16x8;
typedef __attribute__((ext_vector_type(8)))  float  f32x8;

#define NROWS 16384
#define DIN   512
#define DOUT  128
#define KB    128   // K elements staged per iteration
#define MBLK  64    // adj rows per block (4 x 16 per wave)

__device__ inline bf16x8 pack_bf16x8(float4 a0, float4 a1) {
    bf16x8 p;
    p[0] = (__bf16)a0.x; p[1] = (__bf16)a0.y; p[2] = (__bf16)a0.z; p[3] = (__bf16)a0.w;
    p[4] = (__bf16)a1.x; p[5] = (__bf16)a1.y; p[6] = (__bf16)a1.z; p[7] = (__bf16)a1.w;
    return p;
}

// ---------------------------------------------------------------------------
// Kernel 1: xT[DOUT][NROWS] (bf16) = (inputs @ W)^T.  Tiny (2 GFLOP).
// ---------------------------------------------------------------------------
__global__ __launch_bounds__(256) void gemm1_xT(const float* __restrict__ inp,
                                                const float* __restrict__ W,
                                                __bf16* __restrict__ xT)
{
    __shared__ __align__(16) __bf16 atile[16 * 128];

    const int tid  = threadIdx.x;
    const int wave = tid >> 5;
    const int lane = tid & 31;
    const int half = lane >> 4;
    const int r    = lane & 15;
    const int m0   = blockIdx.x * 16;
    const int n0   = wave * 16;

    f32x8 acc = {};

    for (int kb = 0; kb < DIN; kb += 128) {
        {
            const int idx = tid * 8;
            const int row = idx >> 7;
            const int col = idx & 127;
            const float4* s =
                reinterpret_cast<const float4*>(inp + (size_t)(m0 + row) * DIN + kb + col);
            *reinterpret_cast<bf16x8*>(&atile[row * 128 + col]) = pack_bf16x8(s[0], s[1]);
        }
        __syncthreads();

#pragma unroll
        for (int ks = 0; ks < 128; ks += 32) {
            bf16x8 alo = *reinterpret_cast<const bf16x8*>(&atile[r * 128 + ks + half * 16]);
            bf16x8 ahi = *reinterpret_cast<const bf16x8*>(&atile[r * 128 + ks + half * 16 + 8]);
            bf16x16 A;
#pragma unroll
            for (int i = 0; i < 8; ++i) { A[i] = alo[i]; A[8 + i] = ahi[i]; }

            const float* wp = W + (size_t)(kb + ks + half * 16) * DOUT + n0 + r;
            bf16x16 B;
#pragma unroll
            for (int v = 0; v < 16; ++v) B[v] = (__bf16)wp[(size_t)v * DOUT];

            acc = __builtin_amdgcn_wmma_f32_16x16x32_bf16(
                false, A, false, B, (short)0, acc, false, false);
        }
        __syncthreads();
    }

    bf16x8 o;
#pragma unroll
    for (int i = 0; i < 8; ++i) o[i] = (__bf16)acc[i];
    *reinterpret_cast<bf16x8*>(xT + (size_t)(n0 + r) * NROWS + m0 + half * 8) = o;
}

// ---------------------------------------------------------------------------
// Kernel 2: out = adj @ x  (x given as xT bf16 [128][16384]).
// 256 blocks; block = 64 adj rows x 128 cols; 8 waves x 4 accumulators.
// ---------------------------------------------------------------------------
__global__ __launch_bounds__(256) void gemm2_out(const float* __restrict__ adj,
                                                 const __bf16* __restrict__ xT,
                                                 float* __restrict__ out)
{
    __shared__ __align__(16) __bf16 buf[2][MBLK * KB];   // 2 x 16 KB bf16

    const int tid  = threadIdx.x;
    const int wave = tid >> 5;
    const int lane = tid & 31;
    const int half = lane >> 4;
    const int r    = lane & 15;
    const int m0   = blockIdx.x * MBLK;
    const int n0   = wave * 16;

    // Staging: each thread owns one row and 32 consecutive K columns.
    const int srow = tid >> 2;
    const int scol = (tid & 3) * 32;
    const float* abase = adj + (size_t)(m0 + srow) * NROWS + scol;

    f32x8 acc[4] = {{}, {}, {}, {}};
    float4 rg[8];

    // Prologue: fetch + convert tile kb=0 into buf[0].
#pragma unroll
    for (int j = 0; j < 8; ++j)
        rg[j] = reinterpret_cast<const float4*>(abase)[j];
    {
        bf16x8* d = reinterpret_cast<bf16x8*>(&buf[0][srow * KB + scol]);
#pragma unroll
        for (int j = 0; j < 4; ++j) d[j] = pack_bf16x8(rg[2 * j], rg[2 * j + 1]);
    }
    __syncthreads();

    int cur = 0;
    for (int kb = 0; kb < NROWS; kb += KB) {
        const bool more = (kb + KB) < NROWS;

        // Issue next tile's global loads before computing (latency overlap).
        if (more) {
            const float4* nb = reinterpret_cast<const float4*>(abase + kb + KB);
#pragma unroll
            for (int j = 0; j < 8; ++j) rg[j] = nb[j];
        }

        const __bf16* lb = buf[cur];
#pragma unroll
        for (int ks = 0; ks < KB; ks += 32) {
            // B fragment from xT: lane<16 K {0..7,16..23}; lane>=16 +8.
            const __bf16* bp = xT + (size_t)(n0 + r) * NROWS + kb + ks + half * 8;
            bf16x8 blo = *reinterpret_cast<const bf16x8*>(bp);
            bf16x8 bhi = *reinterpret_cast<const bf16x8*>(bp + 16);
            bf16x16 B;
#pragma unroll
            for (int i = 0; i < 8; ++i) { B[i] = blo[i]; B[8 + i] = bhi[i]; }

#pragma unroll
            for (int ms = 0; ms < 4; ++ms) {
                // A fragment: lane<16 K {0..7,16..23}; lane>=16 +8.
                const __bf16* ap = lb + (ms * 16 + r) * KB + ks + half * 8;
                bf16x8 alo = *reinterpret_cast<const bf16x8*>(ap);
                bf16x8 ahi = *reinterpret_cast<const bf16x8*>(ap + 16);
                bf16x16 A;
#pragma unroll
                for (int i = 0; i < 8; ++i) { A[i] = alo[i]; A[8 + i] = ahi[i]; }

                acc[ms] = __builtin_amdgcn_wmma_f32_16x16x32_bf16(
                    false, A, false, B, (short)0, acc[ms], false, false);
            }
        }

        // Convert prefetched tile into the other buffer, then one barrier.
        if (more) {
            bf16x8* d = reinterpret_cast<bf16x8*>(&buf[cur ^ 1][srow * KB + scol]);
#pragma unroll
            for (int j = 0; j < 4; ++j) d[j] = pack_bf16x8(rg[2 * j], rg[2 * j + 1]);
        }
        __syncthreads();
        cur ^= 1;
    }

    // D layout: VGPR i -> row m0 + ms*16 + half*8 + i, col n0 + r.
#pragma unroll
    for (int ms = 0; ms < 4; ++ms) {
        float* op = out + (size_t)(m0 + ms * 16 + half * 8) * DOUT + n0 + r;
#pragma unroll
        for (int i = 0; i < 8; ++i) op[(size_t)i * DOUT] = acc[ms][i];
    }
}

extern "C" void kernel_launch(void* const* d_in, const int* in_sizes, int n_in,
                              void* d_out, int out_size, void* d_ws, size_t ws_size,
                              hipStream_t stream) {
    const float* inputs  = (const float*)d_in[0];   // [16384][512]   f32
    const float* adj     = (const float*)d_in[1];   // [16384][16384] f32
    const float* weights = (const float*)d_in[2];   // [512][128]     f32
    float*       out     = (float*)d_out;           // [16384][128]   f32
    __bf16*      xT      = (__bf16*)d_ws;           // [128][16384]   bf16 (4 MB)

    (void)in_sizes; (void)n_in; (void)out_size; (void)ws_size;

    gemm1_xT<<<dim3(NROWS / 16), dim3(256), 0, stream>>>(inputs, weights, xT);
    gemm2_out<<<dim3(NROWS / MBLK), dim3(256), 0, stream>>>(adj, xT, out);
}
